// HierarchicalSynthesisNet_71339406787371
// MI455X (gfx1250) — matrix operand order repaired and checked by reading, not compiled, not run on hardware
//
#include <hip/hip_runtime.h>
#include <cstdint>
#include <cstddef>

// ---------------------------------------------------------------------------
// Helpers
// ---------------------------------------------------------------------------
#define DEV __device__ __forceinline__

DEV float lrelu02(float v) { return v > 0.f ? v : 0.2f * v; }

DEV unsigned short f2bf(float f) {
    union { float f; unsigned int u; } x; x.f = f;
    unsigned int u = x.u;
    u += 0x7FFFu + ((u >> 16) & 1u);           // round-to-nearest-even
    return (unsigned short)(u >> 16);
}
DEV float bf2f(unsigned short h) {
    union { unsigned int u; float f; } x; x.u = ((unsigned int)h) << 16;
    return x.f;
}

typedef __attribute__((ext_vector_type(16))) __bf16 v16bf;
typedef __attribute__((ext_vector_type(8)))  float  v8f;
union ABfrag { unsigned short u[16]; v16bf v; };

// ---------------------------------------------------------------------------
// Mapping network: adaptive-avg-pool + 6x FC(+LeakyReLU)
// ---------------------------------------------------------------------------
__global__ void pool_mean_kernel(const float* __restrict__ z, float* __restrict__ out, int HW) {
    int idx = blockIdx.x * blockDim.x + threadIdx.x;   // over B*64 = 4096
    if (idx >= 64 * 64) return;
    const float* p = z + (size_t)idx * HW;
    float s = 0.f;
    for (int i = 0; i < HW; ++i) s += p[i];
    out[idx] = s / (float)HW;
}

__global__ void fc_lrelu_kernel(const float* __restrict__ in, const float* __restrict__ W,
                                const float* __restrict__ bias, float* __restrict__ out,
                                int ind, int outd) {
    int idx = blockIdx.x * blockDim.x + threadIdx.x;   // over B*outd
    if (idx >= 64 * outd) return;
    int b = idx / outd, o = idx - b * outd;
    const float* ip = in + (size_t)b * ind;
    const float* wp = W + (size_t)o * ind;
    float s = bias[o];
    for (int i = 0; i < ind; ++i) s += ip[i] * wp[i];
    out[idx] = lrelu02(s);
}

__global__ void add_kernel(const float* __restrict__ a, const float* __restrict__ b,
                           float* __restrict__ o, int n) {
    int idx = blockIdx.x * blockDim.x + threadIdx.x;
    if (idx < n) o[idx] = a[idx] + b[idx];
}

// ---------------------------------------------------------------------------
// proj: conv1x1(z_t) -> lrelu -> instance-norm, out bf16 [B,512,64]
// ---------------------------------------------------------------------------
__global__ __launch_bounds__(64)
void proj_inorm_kernel(const float* __restrict__ zt, const float* __restrict__ w,
                       const float* __restrict__ bias, unsigned short* __restrict__ xout) {
    int oc = blockIdx.x;            // 512
    int b  = blockIdx.y;            // 64
    int p  = threadIdx.x;           // 64 pixels
    const float* zp = zt + (size_t)b * 64 * 64;   // [ic=64][px=64]
    const float* wp = w + (size_t)oc * 64;
    float s = bias[oc];
    for (int i = 0; i < 64; ++i) s += zp[i * 64 + p] * wp[i];
    float v = lrelu02(s);
    __shared__ float r1[64], r2[64];
    r1[p] = v; r2[p] = v * v;
    __syncthreads();
    for (int st = 32; st > 0; st >>= 1) {
        if (p < st) { r1[p] += r1[p + st]; r2[p] += r2[p + st]; }
        __syncthreads();
    }
    float mean = r1[0] * (1.f / 64.f);
    float var  = r2[0] * (1.f / 64.f) - mean * mean;
    float nv   = (v - mean) * rsqrtf(var + 1e-5f);
    xout[((size_t)b * 512 + oc) * 64 + p] = f2bf(nv);
}

// ---------------------------------------------------------------------------
// Per-conv precompute: style s[b,ic], bf16 weights (scaled, tap-major K),
// sum-of-squares weights, demod d[b,oc]
// ---------------------------------------------------------------------------
__global__ void style_kernel(const float* __restrict__ wlat, const float* __restrict__ mw,
                             const float* __restrict__ mb, float* __restrict__ s, int ic) {
    int idx = blockIdx.x * blockDim.x + threadIdx.x;
    if (idx >= 64 * ic) return;
    int b = idx / ic, i = idx - b * ic;
    const float* wp = wlat + (size_t)b * 512;
    const float* mp = mw + (size_t)i * 512;
    float acc = mb[i];
    for (int j = 0; j < 512; ++j) acc += wp[j] * mp[j];
    s[idx] = acc;
}

__global__ void prepw_kernel(const float* __restrict__ wraw, unsigned short* __restrict__ wbf,
                             int ic, int oc) {
    int idx = blockIdx.x * blockDim.x + threadIdx.x;
    int n = oc * ic * 9;
    if (idx >= n) return;
    int t = idx % 9; int rest = idx / 9; int c = rest % ic; int o = rest / ic;
    float scale = rsqrtf((float)(ic * 9));
    wbf[(size_t)o * ic * 9 + (size_t)t * ic + c] = f2bf(wraw[idx] * scale);
}

__global__ void prepwsq_kernel(const float* __restrict__ wraw, float* __restrict__ wsq,
                               int ic, int oc) {
    int idx = blockIdx.x * blockDim.x + threadIdx.x;
    if (idx >= oc * ic) return;
    const float* p = wraw + (size_t)idx * 9;
    float s = 0.f;
    for (int t = 0; t < 9; ++t) s += p[t] * p[t];
    wsq[idx] = s * (1.f / (float)(ic * 9));
}

__global__ void demod_kernel(const float* __restrict__ wsq, const float* __restrict__ s,
                             float* __restrict__ d, int ic, int oc) {
    int idx = blockIdx.x * blockDim.x + threadIdx.x;
    if (idx >= 64 * oc) return;
    int b = idx / oc, o = idx - b * oc;
    const float* wp = wsq + (size_t)o * ic;
    const float* sp = s + (size_t)b * ic;
    float acc = 0.f;
    for (int i = 0; i < ic; ++i) { float sv = sp[i]; acc += wp[i] * sv * sv; }
    d[idx] = rsqrtf(acc + 1e-8f);
}

// ---------------------------------------------------------------------------
// Modulated 3x3 conv via WMMA implicit GEMM, software-pipelined.
//   block = 128 threads = 4 waves. Block tile: 64 oc x 32 pixels.
//   Each wave: 16 oc x 32 px (two accumulators sharing one A fragment).
//   Flat K-chunk loop (k0 = chunk*32 since ic is a power of two). Per-(tap,n)
//   gather descriptors built once in LDS; producer reloads them to registers
//   at tap boundaries. Double-buffered LDS B tile -> ONE barrier per chunk:
//     barrier -> load A + read B-frags(cur) -> stage next(other) -> 2x WMMA
//   UP=1 fuses the bilinear 2x upsample of the input into the staging.
// ---------------------------------------------------------------------------
template <int UP>
__global__ __launch_bounds__(128)
void modconv3x3_wmma_kernel(const unsigned short* __restrict__ xin,
                            const unsigned short* __restrict__ wbf,
                            const float* __restrict__ s,
                            const float* __restrict__ dmod,
                            const float* __restrict__ noise,
                            const float* __restrict__ nsp,
                            unsigned short* __restrict__ xout,
                            int R, int ic, int oc) {
    __shared__ float          s_sh[512];
    __shared__ unsigned short Bt[2][32 * 32];    // double-buffered [n][k]
    __shared__ int            s_off[288][4];     // per (tap, n) gather offsets
    __shared__ float          s_wgt[288][2];     // per (tap, n) bilinear wy, wx

    const int tid  = threadIdx.x;
    const int wave = tid >> 5, lane = tid & 31;
    const int half = lane >> 4, col = lane & 15;
    const int kk   = tid & 31;                   // staged K index within chunk
    const int p0   = blockIdx.x * 32;
    const int ocb  = blockIdx.y * 64;
    const int b    = blockIdx.z;
    const int P    = R * R;
    const int Rin  = UP ? (R >> 1) : R;

    for (int i = tid; i < ic; i += 128) s_sh[i] = s[(size_t)b * ic + i];

    // ---- build gather descriptors once: 9 taps x 32 pixels ----
    for (int e = tid; e < 288; e += 128) {
        int t = e >> 5, n = e & 31;
        int ky = t / 3 - 1, kx = t - (t / 3) * 3 - 1;
        int p = p0 + n;
        int sy = p / R + ky, sx = p % R + kx;
        if ((unsigned)sy >= (unsigned)R || (unsigned)sx >= (unsigned)R) {
            s_off[e][0] = -1; s_off[e][1] = 0; s_off[e][2] = 0; s_off[e][3] = 0;
            s_wgt[e][0] = 0.f; s_wgt[e][1] = 0.f;
        } else if (!UP) {
            int o = sy * R + sx;
            s_off[e][0] = o; s_off[e][1] = o; s_off[e][2] = o; s_off[e][3] = o;
            s_wgt[e][0] = 1.f; s_wgt[e][1] = 1.f;
        } else {
            int my = sy >> 1; int y0 = (sy & 1) ? my : my - 1;
            float wy = (sy & 1) ? 0.75f : 0.25f;
            int y1 = y0 + 1; if (y0 < 0) y0 = 0; if (y1 > Rin - 1) y1 = Rin - 1;
            int mx = sx >> 1; int x0 = (sx & 1) ? mx : mx - 1;
            float wx = (sx & 1) ? 0.75f : 0.25f;
            int x1 = x0 + 1; if (x0 < 0) x0 = 0; if (x1 > Rin - 1) x1 = Rin - 1;
            s_off[e][0] = y0 * Rin + x0; s_off[e][1] = y0 * Rin + x1;
            s_off[e][2] = y1 * Rin + x0; s_off[e][3] = y1 * Rin + x1;
            s_wgt[e][0] = wy; s_wgt[e][1] = wx;
        }
    }
    __syncthreads();

    const int nK      = ic * 9;
    const int cpt     = ic >> 5;                      // chunks per tap (pow2)
    const int cptMask = cpt - 1;
    const int lgCpt   = 31 - __builtin_clz(cpt);
    const int nChunks = 9 * cpt;
    const unsigned short* xb = xin + (size_t)b * ic * Rin * Rin;
    const unsigned short* wrow = wbf + (size_t)(ocb + wave * 16 + col) * nK;  // A row M = lane&15

    // producer-side descriptor registers (tap of the chunk being staged)
    int   off_r[8][4];
    float wgt_r[8][2];
    auto descLoad = [&](int t) {
#pragma unroll
        for (int j = 0; j < 8; ++j) {
            int e = (t << 5) + wave + 4 * j;
            off_r[j][0] = s_off[e][0]; off_r[j][1] = s_off[e][1];
            off_r[j][2] = s_off[e][2]; off_r[j][3] = s_off[e][3];
            wgt_r[j][0] = s_wgt[e][0]; wgt_r[j][1] = s_wgt[e][1];
        }
    };
    auto stageChunk = [&](int j, unsigned short* dst) {
        int cc = j & cptMask;
        int c  = (cc << 5) + kk;
        float sv = s_sh[c];
        const unsigned short* row = xb + (size_t)c * Rin * Rin;
#pragma unroll
        for (int q = 0; q < 8; ++q) {
            int n = wave + 4 * q;
            float v = 0.f;
            int o0 = off_r[q][0];
            if (o0 >= 0) {
                if (!UP) {
                    v = bf2f(row[o0]) * sv;
                } else {
                    float wy = wgt_r[q][0], wx = wgt_r[q][1];
                    float v00 = bf2f(row[o0]);
                    float v01 = bf2f(row[off_r[q][1]]);
                    float v10 = bf2f(row[off_r[q][2]]);
                    float v11 = bf2f(row[off_r[q][3]]);
                    v = (wy * (wx * v00 + (1.f - wx) * v01)
                       + (1.f - wy) * (wx * v10 + (1.f - wx) * v11)) * sv;
                }
            }
            dst[(n << 5) + kk] = f2bf(v);
        }
    };

    v8f acc0 = {0.f, 0.f, 0.f, 0.f, 0.f, 0.f, 0.f, 0.f};
    v8f acc1 = {0.f, 0.f, 0.f, 0.f, 0.f, 0.f, 0.f, 0.f};

    // ---- pipeline prologue: stage chunk 0 ----
    descLoad(0);
    stageChunk(0, Bt[0]);

    for (int i = 0; i < nChunks; ++i) {
        const unsigned short* cur = Bt[i & 1];
        unsigned short*       nxt = Bt[(i + 1) & 1];
        __syncthreads();                           // staging of chunk i visible

        // ---- A fragment for chunk i (k0 = i*32): issue early ----
        int k0 = i << 5;
        ABfrag a, b0, b1;
#pragma unroll
        for (int vg = 0; vg < 8; ++vg) {
            int kOff = (half << 4) + (vg << 1);
            const unsigned short* sp2 = wrow + k0 + kOff;
            a.u[2 * vg]     = sp2[0];
            a.u[2 * vg + 1] = sp2[1];
        }
        // ---- B fragments from current buffer ----
#pragma unroll
        for (int j = 0; j < 8; ++j) {
            int kb = (half << 4) + (j << 1);
            b0.u[2 * j]     = cur[(col << 5) + kb];
            b0.u[2 * j + 1] = cur[(col << 5) + kb + 1];
            b1.u[2 * j]     = cur[((col + 16) << 5) + kb];
            b1.u[2 * j + 1] = cur[((col + 16) << 5) + kb + 1];
        }
        // ---- stage chunk i+1 into the other buffer (overlaps WMMA below) ----
        if (i + 1 < nChunks) {
            if (((i + 1) & cptMask) == 0) descLoad((i + 1) >> lgCpt);
            stageChunk(i + 1, nxt);
        }
        // ---- 2 WMMAs sharing one A fragment ----
        acc0 = __builtin_amdgcn_wmma_f32_16x16x32_bf16(false, a.v, false, b0.v,
                                                       (short)0, acc0, false, false);
        acc1 = __builtin_amdgcn_wmma_f32_16x16x32_bf16(false, a.v, false, b1.v,
                                                       (short)0, acc1, false, false);
    }

    // ---- epilogue: demod + noise + LeakyReLU for both pixel groups ----
    float nsv = nsp[0];
    int p_a = p0 + col, p_b = p0 + 16 + col;
    float nz0 = noise[(size_t)b * P + p_a] * nsv;
    float nz1 = noise[(size_t)b * P + p_b] * nsv;
#pragma unroll
    for (int r = 0; r < 8; ++r) {
        int o = ocb + wave * 16 + (half << 3) + r;    // C layout: M = r + 8*half
        float dv = dmod[(size_t)b * oc + o];
        float y0 = lrelu02(acc0[r] * dv + nz0);
        float y1 = lrelu02(acc1[r] * dv + nz1);
        xout[((size_t)b * oc + o) * P + p_a] = f2bf(y0);
        xout[((size_t)b * oc + o) * P + p_b] = f2bf(y1);
    }
}

// ---------------------------------------------------------------------------
// Bilinear 2x upsample of bf16 x (+ optional conv1x1 injection from z)
// ---------------------------------------------------------------------------
__global__ void upsample_inj_kernel(const unsigned short* __restrict__ xin,
                                    unsigned short* __restrict__ xout,
                                    const float* __restrict__ zinj,
                                    const float* __restrict__ iw,
                                    const float* __restrict__ ib,
                                    int Rin, int C, int hasInj) {
    int Rout = Rin * 2; int P = Rout * Rout;
    size_t total = (size_t)64 * C * P;
    size_t idx = (size_t)blockIdx.x * blockDim.x + threadIdx.x;
    if (idx >= total) return;
    int p = (int)(idx % P);
    size_t bc = idx / P;
    int c = (int)(bc % C); int b = (int)(bc / C);
    int oy = p / Rout, ox = p % Rout;

    int my = oy >> 1; int y0 = (oy & 1) ? my : my - 1;
    float wy = (oy & 1) ? 0.75f : 0.25f;
    int y1 = y0 + 1; if (y0 < 0) y0 = 0; if (y1 > Rin - 1) y1 = Rin - 1;
    int mx = ox >> 1; int x0 = (ox & 1) ? mx : mx - 1;
    float wx = (ox & 1) ? 0.75f : 0.25f;
    int x1 = x0 + 1; if (x0 < 0) x0 = 0; if (x1 > Rin - 1) x1 = Rin - 1;

    const unsigned short* xp = xin + ((size_t)b * C + c) * Rin * Rin;
    float v = wy * (wx * bf2f(xp[y0 * Rin + x0]) + (1.f - wx) * bf2f(xp[y0 * Rin + x1]))
            + (1.f - wy) * (wx * bf2f(xp[y1 * Rin + x0]) + (1.f - wx) * bf2f(xp[y1 * Rin + x1]));

    if (hasInj) {
        const float* zp = zinj + (size_t)b * 64 * P + p;
        const float* wp = iw + (size_t)c * 64;
        float a = ib[c];
        for (int i = 0; i < 64; ++i) a += zp[(size_t)i * P] * wp[i];
        v += a;
    }
    xout[idx] = f2bf(v);
}

__global__ void upsample_rgb_kernel(const float* __restrict__ in, float* __restrict__ out, int Rin) {
    int Rout = Rin * 2; int P = Rout * Rout;
    size_t total = (size_t)64 * 3 * P;
    size_t idx = (size_t)blockIdx.x * blockDim.x + threadIdx.x;
    if (idx >= total) return;
    int p = (int)(idx % P);
    size_t bc = idx / P;
    int c = (int)(bc % 3); int b = (int)(bc / 3);
    int oy = p / Rout, ox = p % Rout;

    int my = oy >> 1; int y0 = (oy & 1) ? my : my - 1;
    float wy = (oy & 1) ? 0.75f : 0.25f;
    int y1 = y0 + 1; if (y0 < 0) y0 = 0; if (y1 > Rin - 1) y1 = Rin - 1;
    int mx = ox >> 1; int x0 = (ox & 1) ? mx : mx - 1;
    float wx = (ox & 1) ? 0.75f : 0.25f;
    int x1 = x0 + 1; if (x0 < 0) x0 = 0; if (x1 > Rin - 1) x1 = Rin - 1;

    const float* xp = in + ((size_t)b * 3 + c) * Rin * Rin;
    out[idx] = wy * (wx * xp[y0 * Rin + x0] + (1.f - wx) * xp[y0 * Rin + x1])
             + (1.f - wy) * (wx * xp[y1 * Rin + x0] + (1.f - wx) * xp[y1 * Rin + x1]);
}

// ---------------------------------------------------------------------------
// RGB 1x1 modulated conv (oc=3, no demod): out = prev + lrelu(conv + noise*ns)
// optional tanh on final stage
// ---------------------------------------------------------------------------
__global__ void rgbconv_kernel(const unsigned short* __restrict__ xin,
                               const float* __restrict__ w,
                               const float* __restrict__ s,
                               const float* __restrict__ noise,
                               const float* __restrict__ nsp,
                               const float* __restrict__ prev,
                               float* __restrict__ out,
                               int P, int ic, int finalTanh) {
    int idx = blockIdx.x * blockDim.x + threadIdx.x;   // over B*P
    if (idx >= 64 * P) return;
    int b = idx / P, p = idx - b * P;
    const unsigned short* xp = xin + (size_t)b * ic * P + p;
    const float* sp = s + (size_t)b * ic;
    float a0 = 0.f, a1 = 0.f, a2 = 0.f;
    for (int c = 0; c < ic; ++c) {
        float xv = bf2f(xp[(size_t)c * P]) * sp[c];
        a0 += w[c] * xv;
        a1 += w[ic + c] * xv;
        a2 += w[2 * ic + c] * xv;
    }
    float scale = rsqrtf((float)ic);
    float nz = noise[idx] * nsp[0];
    float v0 = lrelu02(a0 * scale + nz);
    float v1 = lrelu02(a1 * scale + nz);
    float v2 = lrelu02(a2 * scale + nz);
    if (prev) {
        v0 += prev[((size_t)b * 3 + 0) * P + p];
        v1 += prev[((size_t)b * 3 + 1) * P + p];
        v2 += prev[((size_t)b * 3 + 2) * P + p];
    }
    if (finalTanh) { v0 = tanhf(v0); v1 = tanhf(v1); v2 = tanhf(v2); }
    out[((size_t)b * 3 + 0) * P + p] = v0;
    out[((size_t)b * 3 + 1) * P + p] = v1;
    out[((size_t)b * 3 + 2) * P + p] = v2;
}

// ---------------------------------------------------------------------------
// Host orchestration
// ---------------------------------------------------------------------------
extern "C" void kernel_launch(void* const* d_in, const int* in_sizes, int n_in,
                              void* d_out, int out_size, void* d_ws, size_t ws_size,
                              hipStream_t stream) {
    (void)in_sizes; (void)n_in; (void)out_size;
    auto F = [&](int i) -> const float* { return (const float*)d_in[i]; };

    // --- workspace layout ---
    char* ws = (char*)d_ws;
    size_t off = 0;
    auto alloc = [&](size_t bytes) -> char* {
        char* p = ws + off;
        off = (off + bytes + 255) & ~(size_t)255;
        return p;
    };
    float* MAPT = (float*)alloc(64 * 64 * 4);
    float* MAPM = (float*)alloc(64 * 64 * 4);
    float* T0   = (float*)alloc(64 * 512 * 4);
    float* T1   = (float*)alloc(64 * 512 * 4);
    float* WT   = (float*)alloc(64 * 512 * 4);
    float* WM   = (float*)alloc(64 * 512 * 4);
    float* WV   = (float*)alloc(64 * 512 * 4);
    float* SB   = (float*)alloc(64 * 512 * 4);
    float* SB2  = (float*)alloc(64 * 512 * 4);
    float* DB   = (float*)alloc(64 * 512 * 4);
    unsigned short* WBF = (unsigned short*)alloc((size_t)512 * 512 * 9 * 2);
    float* WSQ  = (float*)alloc((size_t)512 * 512 * 4);
    float* RGBA = (float*)alloc((size_t)64 * 3 * 128 * 128 * 4);
    float* RGBB = (float*)alloc((size_t)64 * 3 * 128 * 128 * 4);
    unsigned short* XA = (unsigned short*)alloc((size_t)64 * 64 * 128 * 128 * 2);  // 134 MB
    unsigned short* XB = (unsigned short*)alloc((size_t)64 * 512 * 32 * 32 * 2);   //  67 MB
    if (off > ws_size) return;   // workspace too small: bail deterministically

    const int TB = 256;
    const int g512 = (64 * 512 + TB - 1) / TB;

    // --- mapping network: w = mlp(z_t) + mlp(z_m) ---
    pool_mean_kernel<<<(4096 + TB - 1) / TB, TB, 0, stream>>>(F(0), MAPT, 64);
    pool_mean_kernel<<<(4096 + TB - 1) / TB, TB, 0, stream>>>(F(1), MAPM, 256);
    fc_lrelu_kernel<<<g512, TB, 0, stream>>>(MAPT, F(3), F(4), T0, 64, 512);
    fc_lrelu_kernel<<<g512, TB, 0, stream>>>(T0, F(5) + 0 * 512 * 512, F(6) + 0 * 512, T1, 512, 512);
    fc_lrelu_kernel<<<g512, TB, 0, stream>>>(T1, F(5) + 1 * 512 * 512, F(6) + 1 * 512, T0, 512, 512);
    fc_lrelu_kernel<<<g512, TB, 0, stream>>>(T0, F(5) + 2 * 512 * 512, F(6) + 2 * 512, T1, 512, 512);
    fc_lrelu_kernel<<<g512, TB, 0, stream>>>(T1, F(5) + 3 * 512 * 512, F(6) + 3 * 512, T0, 512, 512);
    fc_lrelu_kernel<<<g512, TB, 0, stream>>>(T0, F(5) + 4 * 512 * 512, F(6) + 4 * 512, WT, 512, 512);
    fc_lrelu_kernel<<<g512, TB, 0, stream>>>(MAPM, F(7), F(8), T0, 64, 512);
    fc_lrelu_kernel<<<g512, TB, 0, stream>>>(T0, F(9) + 0 * 512 * 512, F(10) + 0 * 512, T1, 512, 512);
    fc_lrelu_kernel<<<g512, TB, 0, stream>>>(T1, F(9) + 1 * 512 * 512, F(10) + 1 * 512, T0, 512, 512);
    fc_lrelu_kernel<<<g512, TB, 0, stream>>>(T0, F(9) + 2 * 512 * 512, F(10) + 2 * 512, T1, 512, 512);
    fc_lrelu_kernel<<<g512, TB, 0, stream>>>(T1, F(9) + 3 * 512 * 512, F(10) + 3 * 512, T0, 512, 512);
    fc_lrelu_kernel<<<g512, TB, 0, stream>>>(T0, F(9) + 4 * 512 * 512, F(10) + 4 * 512, WM, 512, 512);
    add_kernel<<<g512, TB, 0, stream>>>(WT, WM, WV, 64 * 512);

    // --- proj: x8 = inorm(lrelu(conv1x1(z_t))) -> XB ---
    proj_inorm_kernel<<<dim3(512, 64), 64, 0, stream>>>(F(0), F(11), F(12), XB);

    auto style = [&](int mwIdx, int mbIdx, int icd, float* dst) {
        style_kernel<<<(64 * icd + TB - 1) / TB, TB, 0, stream>>>(WV, F(mwIdx), F(mbIdx), dst, icd);
    };
    auto conv = [&](int base, const unsigned short* xi, unsigned short* xo,
                    int R, int icd, int ocd, int up, int nzIdx) {
        prepw_kernel<<<((ocd * icd * 9) + TB - 1) / TB, TB, 0, stream>>>(F(base), WBF, icd, ocd);
        prepwsq_kernel<<<((ocd * icd) + TB - 1) / TB, TB, 0, stream>>>(F(base), WSQ, icd, ocd);
        style(base + 1, base + 2, icd, SB);
        demod_kernel<<<((64 * ocd) + TB - 1) / TB, TB, 0, stream>>>(WSQ, SB, DB, icd, ocd);
        dim3 g((R * R) / 32, ocd / 64, 64);
        if (up)
            modconv3x3_wmma_kernel<1><<<g, 128, 0, stream>>>(
                xi, WBF, SB, DB, F(nzIdx), F(base + 3), xo, R, icd, ocd);
        else
            modconv3x3_wmma_kernel<0><<<g, 128, 0, stream>>>(
                xi, WBF, SB, DB, F(nzIdx), F(base + 3), xo, R, icd, ocd);
    };
    auto rgb = [&](int base, const unsigned short* xi, int P, int icd,
                   const float* prev, float* outp, int nzIdx, int fin) {
        style(base + 1, base + 2, icd, SB2);
        rgbconv_kernel<<<((64 * P) + TB - 1) / TB, TB, 0, stream>>>(
            xi, F(base), SB2, F(nzIdx), F(base + 3), prev, outp, P, icd, fin);
    };
    auto upx = [&](const unsigned short* xi, unsigned short* xo, int Rin, int C,
                   const float* z, const float* iw, const float* ibias) {
        size_t tot = (size_t)64 * C * 4 * Rin * Rin;
        upsample_inj_kernel<<<(unsigned)((tot + TB - 1) / TB), TB, 0, stream>>>(
            xi, xo, z, iw, ibias, Rin, C, 1);
    };
    auto uprgb = [&](const float* i, float* o, int Rin) {
        size_t tot = (size_t)64 * 3 * 4 * Rin * Rin;
        upsample_rgb_kernel<<<(unsigned)((tot + TB - 1) / TB), TB, 0, stream>>>(i, o, Rin);
    };

    // --- synthesis pipeline ---
    conv(17, XB, XA, 8, 512, 512, 0, 57);              // c8:  x8  -> x8c (XA)
    rgb(21, XA, 64, 512, nullptr, RGBA, 58, 0);        // r8

    upx(XA, XB, 8, 512, F(1), F(13), F(14));           // x16u = up(x8c)+inj_mid(z_m)
    uprgb(RGBA, RGBB, 8);
    conv(25, XB, XA, 16, 512, 512, 0, 59);             // c16
    rgb(29, XA, 256, 512, RGBB, RGBA, 60, 0);          // r16

    upx(XA, XB, 16, 512, F(2), F(15), F(16));          // x32u = up(x16c)+inj_bot(z_b)
    uprgb(RGBA, RGBB, 16);
    conv(33, XB, XA, 32, 512, 256, 0, 61);             // c32
    rgb(37, XA, 1024, 256, RGBB, RGBA, 62, 0);         // r32

    uprgb(RGBA, RGBB, 32);
    conv(41, XA, XB, 64, 256, 128, 1, 63);             // c64, fused 2x up of x32c
    rgb(45, XB, 4096, 128, RGBB, RGBA, 64, 0);         // r64

    uprgb(RGBA, RGBB, 64);
    conv(49, XB, XA, 128, 128, 64, 1, 65);             // c128, fused 2x up of x64c
    rgb(53, XA, 16384, 64, RGBB, (float*)d_out, 66, 1); // r128 + tanh -> d_out
}
